// AttentionBlock_16776142258690
// MI455X (gfx1250) — matrix operand order repaired and checked by reading, not compiled, not run on hardware
//
#include <hip/hip_runtime.h>

// ---------------------------------------------------------------------------
// CDNA5 (gfx1250) attention block: bf16 WMMA + async global->LDS staging.
// ---------------------------------------------------------------------------

typedef __bf16 v16bf __attribute__((ext_vector_type(16)));
typedef float  v8f   __attribute__((ext_vector_type(8)));

struct U4 { unsigned int x, y, z, w; };          // POD 16-byte chunk
union Frag { unsigned short u[16]; U4 q[2]; v16bf v; };

__device__ __forceinline__ unsigned short f2bf(float f) {
    unsigned int u = __float_as_uint(f);
    u += 0x7FFFu + ((u >> 16) & 1u);             // round-to-nearest-even
    return (unsigned short)(u >> 16);
}
__device__ __forceinline__ float bf2f(unsigned short h) {
    return __uint_as_float(((unsigned int)h) << 16);
}

// LDS byte offset of a __shared__ pointer (AS3 pointers are 32-bit LDS offsets)
#define LDS_OFF(p) ((unsigned)(unsigned long long)(__attribute__((address_space(3))) void*)(p))

// gfx1250 async global->LDS 16-byte copy (per-lane, tracked by ASYNCcnt)
__device__ __forceinline__ void async_ld16(unsigned lds, const void* g) {
    asm volatile("global_load_async_to_lds_b128 %0, %1, off"
                 :: "v"(lds), "v"((unsigned long long)g)
                 : "memory");
}
__device__ __forceinline__ void async_fence() {
    asm volatile("s_wait_asynccnt 0x0" ::: "memory");
}

// ---------------------------------------------------------------------------
// f32 -> bf16 conversion (weights / text)
// ---------------------------------------------------------------------------
__global__ void k_f2bf(const float* __restrict__ s, unsigned short* __restrict__ d,
                       long long n) {
    long long i = (long long)blockIdx.x * blockDim.x + threadIdx.x;
    if (i < n) d[i] = f2bf(s[i]);
}

// ---------------------------------------------------------------------------
// GroupNorm(32, C=256, eps=1e-6) over (B,C,H,W) -> bf16 (B,C,HW)
// one block per (b, group): 8 channels * 4096 = 32768 elems
// ---------------------------------------------------------------------------
__global__ __launch_bounds__(256)
void k_groupnorm(const float* __restrict__ x, const float* __restrict__ g,
                 const float* __restrict__ bt, unsigned short* __restrict__ y) {
    int bg = blockIdx.x, b = bg >> 5, grp = bg & 31, tid = threadIdx.x;
    const float* xp = x + ((long long)b * 256 + grp * 8) * 4096;
    unsigned short* yp = y + ((long long)b * 256 + grp * 8) * 4096;
    float s = 0.f, ss = 0.f;
    for (int i = tid; i < 32768; i += 256) { float v = xp[i]; s += v; ss += v * v; }
    __shared__ float rs[256], rq[256];
    rs[tid] = s; rq[tid] = ss; __syncthreads();
    for (int st = 128; st > 0; st >>= 1) {
        if (tid < st) { rs[tid] += rs[tid + st]; rq[tid] += rq[tid + st]; }
        __syncthreads();
    }
    float mean = rs[0] * (1.f / 32768.f);
    float var  = rq[0] * (1.f / 32768.f) - mean * mean;
    float rstd = rsqrtf(var + 1e-6f);
    for (int i = tid; i < 32768; i += 256) {
        int c = grp * 8 + (i >> 12);
        yp[i] = f2bf((xp[i] - mean) * rstd * g[c] + bt[c]);
    }
}

// ---------------------------------------------------------------------------
// LayerNorm over rows of 256, f32 in -> bf16 out. grid = #rows.
// ---------------------------------------------------------------------------
__global__ __launch_bounds__(256)
void k_layernorm(const float* __restrict__ x, const float* __restrict__ g,
                 const float* __restrict__ bt, unsigned short* __restrict__ y,
                 float eps) {
    long long row = blockIdx.x;
    int tid = threadIdx.x;
    float v = x[row * 256 + tid];
    __shared__ float rs[256], rq[256];
    rs[tid] = v; rq[tid] = v * v; __syncthreads();
    for (int st = 128; st > 0; st >>= 1) {
        if (tid < st) { rs[tid] += rs[tid + st]; rq[tid] += rq[tid + st]; }
        __syncthreads();
    }
    float mean = rs[0] * (1.f / 256.f);
    float var  = rq[0] * (1.f / 256.f) - mean * mean;
    float o = (v - mean) * rsqrtf(var + eps) * g[tid] + bt[tid];
    y[row * 256 + tid] = f2bf(o);
}

// ---------------------------------------------------------------------------
// GeGLU: h (ntok,2048) bf16 -> out (ntok,1024) bf16, exact gelu
// ---------------------------------------------------------------------------
__global__ void k_geglu(const unsigned short* __restrict__ h,
                        unsigned short* __restrict__ o, long long total) {
    long long i = (long long)blockIdx.x * blockDim.x + threadIdx.x;
    if (i >= total) return;
    long long t = i >> 10; int j = (int)(i & 1023);
    float a = bf2f(h[t * 2048 + j]);
    float g = bf2f(h[t * 2048 + 1024 + j]);
    float ge = 0.5f * g * (1.f + erff(g * 0.70710678118654752f));
    o[i] = f2bf(a * ge);
}

// ---------------------------------------------------------------------------
// bf16 WMMA GEMM: out = A(MxK) * B(KxN) [+bias][+f32 residual]
// A row-major bf16. B: transB=0 -> (K,N) row-major; transB=1 -> (N,K) row-major.
// Block: 256 threads = 8 waves; tile 128(M) x 64(N) x 32(K).
// A tile and transB=1 B tile staged with global_load_async_to_lds_b128.
// ---------------------------------------------------------------------------
#define BM 128
#define BN 64
#define BK 32
#define LDS_PAD 48

__global__ __launch_bounds__(256)
void k_gemm(const unsigned short* __restrict__ A, long long sAb, int lda,
            const unsigned short* __restrict__ Bm, long long sBb, int ldb, int transB,
            float* outF, unsigned short* outH, long long sOb, int ldo,
            const float* __restrict__ bias, int biasMode,
            const float* __restrict__ resid, long long sRb,
            int M, int N, int K) {
    __shared__ __align__(16) unsigned short As[BM][LDS_PAD];
    __shared__ __align__(16) unsigned short Bs[BN][LDS_PAD];   // [n][k]
    int b   = blockIdx.z;
    A  += (long long)b * sAb;
    Bm += (long long)b * sBb;
    long long obase = (long long)b * sOb;
    int m0 = blockIdx.y * BM, n0 = blockIdx.x * BN;
    int tid = threadIdx.x, lane = tid & 31, w = tid >> 5, hl = lane >> 4;

    v8f acc[4];
#pragma unroll
    for (int j = 0; j < 4; ++j)
#pragma unroll
        for (int e = 0; e < 8; ++e) acc[j][e] = 0.f;

    int rowA = tid >> 1;                 // 0..127
    int colA = (tid & 1) * 16;           // 0 or 16

    for (int k0 = 0; k0 < K; k0 += BK) {
        // ---- stage A tile (128x32): async copy straight into LDS ----
        const unsigned short* ap = A + (long long)(m0 + rowA) * lda + k0 + colA;
        if (m0 + rowA < M) {
            unsigned la = LDS_OFF(&As[rowA][colA]);
            async_ld16(la,      ap);
            async_ld16(la + 16, ap + 8);
        } else {
#pragma unroll
            for (int e = 0; e < 16; ++e) As[rowA][colA + e] = 0;
        }
        // ---- stage B tile (32x64) into Bs[n][k] ----
        if (!transB) {
            int kk = tid >> 3;           // 0..31
            int nn = (tid & 7) * 8;      // 0..56
            const unsigned short* bp = Bm + (long long)(k0 + kk) * ldb + n0 + nn;
            U4 pv = *(const U4*)bp;
            const unsigned short* pu = (const unsigned short*)&pv;
#pragma unroll
            for (int e = 0; e < 8; ++e) Bs[nn + e][kk] = pu[e];
        } else {
            int nn = tid >> 2;           // 0..63
            int kk = (tid & 3) * 8;      // 0,8,16,24
            const unsigned short* bp = Bm + (long long)(n0 + nn) * ldb + k0 + kk;
            async_ld16(LDS_OFF(&Bs[nn][kk]), bp);
        }
        async_fence();                   // ASYNCcnt -> 0 before the barrier
        __syncthreads();

        // prefetch next K tile (gfx1250 global_prefetch_b8)
        if (k0 + BK < K) {
            __builtin_prefetch(A + (long long)(m0 + rowA) * lda + k0 + BK + colA, 0, 1);
            __builtin_prefetch(Bm + (transB ? ((long long)(n0 + (tid >> 2)) * ldb + k0 + BK)
                                            : ((long long)(k0 + BK + (tid >> 3)) * ldb + n0)),
                               0, 1);
        }

        // ---- fragments + WMMA ----
        Frag af;
        int mloc = w * 16 + (lane & 15);
        af.q[0] = *(const U4*)&As[mloc][hl * 8];
        af.q[1] = *(const U4*)&As[mloc][16 + hl * 8];
#pragma unroll
        for (int j = 0; j < 4; ++j) {
            Frag bf;
            int nloc = j * 16 + (lane & 15);
            bf.q[0] = *(const U4*)&Bs[nloc][hl * 8];
            bf.q[1] = *(const U4*)&Bs[nloc][16 + hl * 8];
            acc[j] = __builtin_amdgcn_wmma_f32_16x16x32_bf16(
                false, af.v, false, bf.v, (short)0, acc[j], false, false);
        }
        __syncthreads();
    }

    // ---- epilogue ----
#pragma unroll
    for (int j = 0; j < 4; ++j) {
        int ncol = n0 + j * 16 + (lane & 15);
        if (ncol >= N) continue;
#pragma unroll
        for (int r = 0; r < 8; ++r) {
            int mrow = m0 + w * 16 + r + 8 * hl;
            if (mrow >= M) continue;
            float v = acc[j][r];
            if (bias)  v += (biasMode == 2) ? bias[mrow] : bias[ncol];
            if (resid) v += resid[(long long)b * sRb + (long long)mrow * ldo + ncol];
            long long off = obase + (long long)mrow * ldo + ncol;
            if (outF) outF[off] = v;
            if (outH) outH[off] = f2bf(v);
        }
    }
}

// ---------------------------------------------------------------------------
// Flash attention: heads=8, dh=32. bf16 Q,K,V (B,T,256) head-major in D.
// Block = 128 threads (4 waves), 64 queries; key chunks of 32.
// K tile staged async to LDS; V transposed via VGPR path; P relaid per-wave.
// ---------------------------------------------------------------------------
__global__ __launch_bounds__(128)
void k_flash(const unsigned short* __restrict__ Q,
             const unsigned short* __restrict__ Kv,
             const unsigned short* __restrict__ Vv,
             unsigned short* __restrict__ O,
             int Tq, int Tk, long long sQ, long long sKV, float scale) {
    __shared__ __align__(16) unsigned short Ks[32][LDS_PAD];      // [key][dh]
    __shared__ __align__(16) unsigned short Vs[32][LDS_PAD];      // [dh][key]
    __shared__ __align__(16) unsigned short Ps[4][16][LDS_PAD];   // per-wave P
    int b = blockIdx.z, h = blockIdx.y;
    int q0 = blockIdx.x * 64;
    int tid = threadIdx.x, lane = tid & 31, w = tid >> 5, hl = lane >> 4;
    int kcol = lane & 15;

    const unsigned short* qb = Q  + (long long)b * sQ  + h * 32;
    const unsigned short* kb = Kv + (long long)b * sKV + h * 32;
    const unsigned short* vb = Vv + (long long)b * sKV + h * 32;

    Frag qf;
    {
        int qRow = q0 + w * 16 + (lane & 15);
        const unsigned short* qp = qb + (long long)qRow * 256;
        qf.q[0] = *(const U4*)(qp + hl * 8);
        qf.q[1] = *(const U4*)(qp + 16 + hl * 8);
    }

    v8f o0, o1; float mrow[8], lrow[8];
#pragma unroll
    for (int e = 0; e < 8; ++e) { o0[e] = 0.f; o1[e] = 0.f; mrow[e] = -1e30f; lrow[e] = 0.f; }

    for (int kc = 0; kc < Tk; kc += 32) {
        __syncthreads();
        {   // stage K,V chunk (32 keys x 32 dims)
            int key = tid >> 2, d0 = (tid & 3) * 8, gkey = kc + key;
            if (gkey < Tk) {
                async_ld16(LDS_OFF(&Ks[key][d0]), kb + (long long)gkey * 256 + d0);
                U4 vv = *(const U4*)(vb + (long long)gkey * 256 + d0);
                const unsigned short* vu = (const unsigned short*)&vv;
#pragma unroll
                for (int e = 0; e < 8; ++e) Vs[d0 + e][key] = vu[e];
            } else {
#pragma unroll
                for (int e = 0; e < 8; ++e) { Ks[key][d0 + e] = 0; Vs[d0 + e][key] = 0; }
            }
        }
        async_fence();
        __syncthreads();

        Frag kf0, kf1;
        kf0.q[0] = *(const U4*)&Ks[kcol][hl * 8];
        kf0.q[1] = *(const U4*)&Ks[kcol][16 + hl * 8];
        kf1.q[0] = *(const U4*)&Ks[16 + kcol][hl * 8];
        kf1.q[1] = *(const U4*)&Ks[16 + kcol][16 + hl * 8];

        v8f s0, s1;
#pragma unroll
        for (int e = 0; e < 8; ++e) { s0[e] = 0.f; s1[e] = 0.f; }
        s0 = __builtin_amdgcn_wmma_f32_16x16x32_bf16(false, qf.v, false, kf0.v, (short)0, s0, false, false);
        s1 = __builtin_amdgcn_wmma_f32_16x16x32_bf16(false, qf.v, false, kf1.v, (short)0, s1, false, false);

        float mask0 = (kc + kcol      < Tk) ? 0.f : -1e30f;
        float mask1 = (kc + 16 + kcol < Tk) ? 0.f : -1e30f;

#pragma unroll
        for (int r = 0; r < 8; ++r) {
            float a0 = s0[r] * scale + mask0;
            float a1 = s1[r] * scale + mask1;
            float mx = fmaxf(a0, a1);
#pragma unroll
            for (int off = 1; off < 16; off <<= 1) mx = fmaxf(mx, __shfl_xor(mx, off, 16));
            float mnew  = fmaxf(mrow[r], mx);
            float alpha = __expf(mrow[r] - mnew);
            float p0 = __expf(a0 - mnew), p1 = __expf(a1 - mnew);
            float ps = p0 + p1;
#pragma unroll
            for (int off = 1; off < 16; off <<= 1) ps += __shfl_xor(ps, off, 16);
            lrow[r] = lrow[r] * alpha + ps;
            mrow[r] = mnew;
            o0[r] *= alpha; o1[r] *= alpha;
            int prow = r + 8 * hl;
            Ps[w][prow][kcol]      = f2bf(p0);
            Ps[w][prow][16 + kcol] = f2bf(p1);
        }

        Frag pf;                         // same-wave LDS round trip (in-order)
        int pm = lane & 15;
        pf.q[0] = *(const U4*)&Ps[w][pm][hl * 8];
        pf.q[1] = *(const U4*)&Ps[w][pm][16 + hl * 8];

        Frag vf0, vf1;
        vf0.q[0] = *(const U4*)&Vs[kcol][hl * 8];
        vf0.q[1] = *(const U4*)&Vs[kcol][16 + hl * 8];
        vf1.q[0] = *(const U4*)&Vs[16 + kcol][hl * 8];
        vf1.q[1] = *(const U4*)&Vs[16 + kcol][16 + hl * 8];

        o0 = __builtin_amdgcn_wmma_f32_16x16x32_bf16(false, pf.v, false, vf0.v, (short)0, o0, false, false);
        o1 = __builtin_amdgcn_wmma_f32_16x16x32_bf16(false, pf.v, false, vf1.v, (short)0, o1, false, false);
    }

#pragma unroll
    for (int r = 0; r < 8; ++r) {
        float inv = lrow[r] > 0.f ? 1.f / lrow[r] : 0.f;
        int qrow = q0 + w * 16 + r + 8 * hl;
        unsigned short* op = O + (long long)b * sQ + (long long)qrow * 256 + h * 32;
        op[kcol]      = f2bf(o0[r] * inv);
        op[16 + kcol] = f2bf(o1[r] * inv);
    }
}

// ---------------------------------------------------------------------------
// Host orchestration
// ---------------------------------------------------------------------------
extern "C" void kernel_launch(void* const* d_in, const int* in_sizes, int n_in,
                              void* d_out, int out_size, void* d_ws, size_t ws_size,
                              hipStream_t stream) {
    (void)in_sizes; (void)n_in; (void)out_size; (void)ws_size;
    typedef unsigned short us;
    typedef long long ll;

    const float* p_img  = (const float*)d_in[0];
    const float* p_text = (const float*)d_in[1];
    const float* p_gng  = (const float*)d_in[2];
    const float* p_gnb  = (const float*)d_in[3];
    const float* p_ciw  = (const float*)d_in[4];
    const float* p_cib  = (const float*)d_in[5];
    const float* p_ln1g = (const float*)d_in[6];
    const float* p_ln1b = (const float*)d_in[7];
    const float* p_a1wq = (const float*)d_in[8];
    const float* p_a1wk = (const float*)d_in[9];
    const float* p_a1wv = (const float*)d_in[10];
    const float* p_a1wo = (const float*)d_in[11];
    const float* p_a1bo = (const float*)d_in[12];
    const float* p_ln2g = (const float*)d_in[13];
    const float* p_ln2b = (const float*)d_in[14];
    const float* p_a2wq = (const float*)d_in[15];
    const float* p_a2wk = (const float*)d_in[16];
    const float* p_a2wv = (const float*)d_in[17];
    const float* p_a2wo = (const float*)d_in[18];
    const float* p_a2bo = (const float*)d_in[19];
    const float* p_ln3g = (const float*)d_in[20];
    const float* p_ln3b = (const float*)d_in[21];
    const float* p_l1w  = (const float*)d_in[22];
    const float* p_l1b  = (const float*)d_in[23];
    const float* p_l2w  = (const float*)d_in[24];
    const float* p_l2b  = (const float*)d_in[25];
    const float* p_cow  = (const float*)d_in[26];
    const float* p_cob  = (const float*)d_in[27];

    const int Bz = 2, T = 4096, HW = 4096;
    const ll NA = (ll)Bz * T * 256;                 // activation elems

    char* ws = (char*)d_ws;
    size_t off = 0;
    auto alloc = [&](size_t bytes) -> void* {
        void* p = ws + off;
        off = (off + bytes + 255) & ~(size_t)255;
        return p;
    };

    us* w_ci = (us*)alloc(65536 * 2);
    us* w_q1 = (us*)alloc(65536 * 2);
    us* w_k1 = (us*)alloc(65536 * 2);
    us* w_v1 = (us*)alloc(65536 * 2);
    us* w_o1 = (us*)alloc(65536 * 2);
    us* w_q2 = (us*)alloc(65536 * 2);
    us* w_k2 = (us*)alloc(196608 * 2);
    us* w_v2 = (us*)alloc(196608 * 2);
    us* w_o2 = (us*)alloc(65536 * 2);
    us* w_l1 = (us*)alloc(524288 * 2);
    us* w_l2 = (us*)alloc(262144 * 2);
    us* w_co = (us*)alloc(65536 * 2);
    us* txtb = (us*)alloc(118272 * 2);
    us* gnb  = (us*)alloc(NA * 2);
    float* convf = (float*)alloc(NA * 4);
    us* x1 = (us*)alloc(NA * 2);
    us* qb = (us*)alloc(NA * 2);
    us* kb = (us*)alloc(NA * 2);
    us* vb = (us*)alloc(NA * 2);
    us* ob = (us*)alloc(NA * 2);
    float* xa = (float*)alloc(NA * 4);
    us* xn = (us*)alloc(NA * 2);
    float* xb = (float*)alloc(NA * 4);
    us* hb = (us*)alloc((ll)Bz * T * 2048 * 2);
    us* gl = (us*)alloc((ll)Bz * T * 1024 * 2);
    us* x3 = (us*)alloc(NA * 2);

    auto cvt = [&](const float* s, us* d, ll n) {
        unsigned int blocks = (unsigned int)((n + 255) / 256);
        k_f2bf<<<blocks, 256, 0, stream>>>(s, d, n);
    };
    cvt(p_ciw, w_ci, 65536);   cvt(p_a1wq, w_q1, 65536);
    cvt(p_a1wk, w_k1, 65536);  cvt(p_a1wv, w_v1, 65536);
    cvt(p_a1wo, w_o1, 65536);  cvt(p_a2wq, w_q2, 65536);
    cvt(p_a2wk, w_k2, 196608); cvt(p_a2wv, w_v2, 196608);
    cvt(p_a2wo, w_o2, 65536);  cvt(p_l1w, w_l1, 524288);
    cvt(p_l2w, w_l2, 262144);  cvt(p_cow, w_co, 65536);
    cvt(p_text, txtb, 118272);

    auto gemm = [&](const us* A, ll sA, int lda, const us* Bm, ll sB, int ldb,
                    int transB, float* oF, us* oH, ll sO, int ldo,
                    const float* bias, int bmode, const float* res, ll sR,
                    int M, int N, int K) {
        dim3 grid((N + BN - 1) / BN, (M + BM - 1) / BM, Bz);
        k_gemm<<<grid, 256, 0, stream>>>(A, sA, lda, Bm, sB, ldb, transB,
                                         oF, oH, sO, ldo, bias, bmode, res, sR, M, N, K);
    };

    // GroupNorm -> bf16
    k_groupnorm<<<Bz * 32, 256, 0, stream>>>(p_img, p_gng, p_gnb, gnb);
    // conv_in: W(256x256) @ gn(256x4096), bias per-row -> f32
    gemm(w_ci, 0, 256, gnb, (ll)256 * HW, HW, 0, convf, nullptr, (ll)256 * HW, HW,
         p_cib, 2, nullptr, 0, 256, HW, 256);
    // LN1 -> bf16 x1 (rows of 256)
    k_layernorm<<<Bz * T, 256, 0, stream>>>(convf, p_ln1g, p_ln1b, x1, 1e-5f);
    // Q,K,V projections
    gemm(x1, (ll)T * 256, 256, w_q1, 0, 256, 0, nullptr, qb, (ll)T * 256, 256,
         nullptr, 0, nullptr, 0, T, 256, 256);
    gemm(x1, (ll)T * 256, 256, w_k1, 0, 256, 0, nullptr, kb, (ll)T * 256, 256,
         nullptr, 0, nullptr, 0, T, 256, 256);
    gemm(x1, (ll)T * 256, 256, w_v1, 0, 256, 0, nullptr, vb, (ll)T * 256, 256,
         nullptr, 0, nullptr, 0, T, 256, 256);
    // self-attention
    const float scale = 0.17677669529663687f;     // 1/sqrt(32)
    k_flash<<<dim3(T / 64, 8, Bz), 128, 0, stream>>>(qb, kb, vb, ob, T, T,
                                                     (ll)T * 256, (ll)T * 256, scale);
    // o-projection (res2 = output)
    gemm(ob, (ll)T * 256, 256, w_o1, 0, 256, 0, xa, nullptr, (ll)T * 256, 256,
         p_a1bo, 1, nullptr, 0, T, 256, 256);
    // LN2
    k_layernorm<<<Bz * T, 256, 0, stream>>>(xa, p_ln2g, p_ln2b, xn, 1e-5f);
    // cross-attention projections
    gemm(xn, (ll)T * 256, 256, w_q2, 0, 256, 0, nullptr, qb, (ll)T * 256, 256,
         nullptr, 0, nullptr, 0, T, 256, 256);
    gemm(txtb, (ll)77 * 768, 768, w_k2, 0, 256, 0, nullptr, kb, (ll)77 * 256, 256,
         nullptr, 0, nullptr, 0, 77, 256, 768);
    gemm(txtb, (ll)77 * 768, 768, w_v2, 0, 256, 0, nullptr, vb, (ll)77 * 256, 256,
         nullptr, 0, nullptr, 0, 77, 256, 768);
    k_flash<<<dim3(T / 64, 8, Bz), 128, 0, stream>>>(qb, kb, vb, ob, T, 77,
                                                     (ll)T * 256, (ll)77 * 256, scale);
    // o-projection + res2 -> xb (res2')
    gemm(ob, (ll)T * 256, 256, w_o2, 0, 256, 0, xb, nullptr, (ll)T * 256, 256,
         p_a2bo, 1, xa, (ll)T * 256, T, 256, 256);
    // LN3
    k_layernorm<<<Bz * T, 256, 0, stream>>>(xb, p_ln3g, p_ln3b, xn, 1e-5f);
    // GeGLU MLP
    gemm(xn, (ll)T * 256, 256, w_l1, 0, 2048, 0, nullptr, hb, (ll)T * 2048, 2048,
         p_l1b, 1, nullptr, 0, T, 2048, 256);
    {
        ll total = (ll)Bz * T * 1024;
        k_geglu<<<(unsigned int)((total + 255) / 256), 256, 0, stream>>>(hb, gl, total);
    }
    gemm(gl, (ll)T * 1024, 1024, w_l2, 0, 256, 0, nullptr, x3, (ll)T * 256, 256,
         p_l2b, 1, xb, (ll)T * 256, T, 256, 1024);
    // conv_out: W(256x256) @ x3^T (transB), bias per-row, + image residual -> d_out f32
    gemm(w_co, 0, 256, x3, (ll)T * 256, 256, 1, (float*)d_out, nullptr,
         (ll)256 * HW, HW, p_cob, 2, p_img, (ll)256 * HW, 256, HW, 256);
}
